// MultiScaleBSQ_60808146977330
// MI455X (gfx1250) — matrix operand order repaired and checked by reading, not compiled, not run on hardware
//
#include <hip/hip_runtime.h>
#include <hip/hip_bf16.h>

typedef __attribute__((ext_vector_type(16))) __bf16 v16bf;
typedef __attribute__((ext_vector_type(8)))  float  v8f;
typedef __attribute__((ext_vector_type(4)))  unsigned int tdm_g0_t;
typedef __attribute__((ext_vector_type(8)))  int          tdm_g1_t;
typedef __attribute__((ext_vector_type(4)))  int          tdm_g4_t;
typedef __attribute__((ext_vector_type(8)))  int          tdm_g8_t;

#define DIM   512
#define CD    32
#define HW    4096
#define BATCH 16
#define NTOK_FULL (BATCH * HW)   // 65536
#define KCHUNK 32

__device__ __forceinline__ float wave_sum(float v) {
#pragma unroll
  for (int off = 16; off > 0; off >>= 1) v += __shfl_xor(v, off, 32);
  return v;
}

// ---------------------------------------------------------------------------
// TDM: DMA a 2D tile (tile1 rows x tile0 elems, fp32, row stride = stride0
// elems) from global memory into LDS at lds_byte_off. D# packing per CDNA5 ISA
// 8.3/8.4: group0 = {count=1 | lds_addr | global_addr(57b) | type=2},
// group1 = {data_size=4B, tensor dims, tile dims, dim0 stride}.
// This toolchain exposes the 6-arg builtin (extra int32x8 group before cpol).
// ---------------------------------------------------------------------------
__device__ __forceinline__ void tdm_load_2d_f32(
    unsigned lds_byte_off, const void* gptr,
    unsigned tile0, unsigned tile1,
    unsigned tensor0, unsigned tensor1,
    unsigned long long stride0_elems)
{
  unsigned long long ga = (unsigned long long)(uintptr_t)gptr;
  tdm_g0_t g0;
  g0.x = 1u;                                             // count=1, user mode
  g0.y = lds_byte_off;                                   // LDS byte address
  g0.z = (unsigned)(ga & 0xFFFFFFFFu);                   // global_addr[31:0]
  g0.w = (unsigned)((ga >> 32) & 0x1FFFFFFu) | (2u << 30); // [56:32] | type=2
  tdm_g1_t g1;
  g1[0] = (int)(2u << 16);                               // data_size=2 (4B)
  g1[1] = (int)((tensor0 & 0xFFFFu) << 16);              // tensor_dim0[15:0]
  g1[2] = (int)((tensor0 >> 16) | ((tensor1 & 0xFFFFu) << 16));
  g1[3] = (int)((tensor1 >> 16) | (tile0 << 16));        // tile_dim0
  g1[4] = (int)tile1;                                    // tile_dim1 (dim2=0)
  g1[5] = (int)(unsigned)(stride0_elems & 0xFFFFFFFFu);  // dim0 stride lo
  g1[6] = (int)(unsigned)((stride0_elems >> 32) & 0xFFFFu);
  g1[7] = 0;
  tdm_g4_t gz4 = {0, 0, 0, 0};
  tdm_g8_t gz8 = {0, 0, 0, 0, 0, 0, 0, 0};
  __builtin_amdgcn_tensor_load_to_lds(g0, g1, gz4, gz4, gz8, 0);
}

// ---------------------------------------------------------------------------
// project_in:  r[tok, d] = sum_k x[b, k, hw] * w_in[k, d] + b_in[d]
// x is channel-major (B, 512, 4096); r is token-major (65536, 32).
// Block = 128 tokens (8 waves x 16-token M tiles). The strided x tile is
// DMA'd to LDS by the TDM (one tensor_load per 32-K chunk); w_in is staged
// once per block in LDS as bf16. Two chained bf16 WMMAs per wave per K-step.
// Also zero-initializes qsum.
// ---------------------------------------------------------------------------
__global__ __launch_bounds__(256) void k_project_in(
    const float* __restrict__ x, const float* __restrict__ w_in,
    const float* __restrict__ b_in, float* __restrict__ r,
    float* __restrict__ qsum)
{
  __shared__ float  xt[KCHUNK][128];    // 16 KB: x tile [k][token]
  __shared__ __bf16 wl[DIM * CD];       // 32 KB: w_in as bf16 [k][n]

  const int lane = threadIdx.x & 31;
  const int wv   = threadIdx.x >> 5;
  const int tok0_blk = blockIdx.x * 128;
  const int b    = tok0_blk >> 12;            // /4096 (never straddles batch)
  const int hw0  = tok0_blk & 4095;
  const int tok0 = tok0_blk + wv * 16;        // this wave's 16-token M tile
  const int m    = lane & 15;
  const int hi8  = (lane >= 16) ? 8 : 0;      // A K-offset AND C M-offset
  const int hi16 = (lane >= 16) ? 16 : 0;     // unused; B K-offset is hi8 as for A

  for (int i = threadIdx.x; i < DIM * CD; i += 256)
    wl[i] = (__bf16)w_in[i];

  const unsigned xt_lds = (unsigned)(uintptr_t)&xt[0][0];

  v8f c0 = {}; v8f c1 = {};
  for (int kb = 0; kb < DIM; kb += KCHUNK) {
    __syncthreads();                          // prev chunk consumed / wl ready
    if (wv == 0) {
      const float* gsrc = x + (((size_t)(b * DIM + kb)) << 12) + hw0;
      tdm_load_2d_f32(xt_lds, gsrc, /*tile0=*/128, /*tile1=*/KCHUNK,
                      /*tensor0=*/128, /*tensor1=*/KCHUNK, /*stride0=*/4096ull);
      __builtin_amdgcn_s_wait_tensorcnt(0);
    }
    __syncthreads();                          // tile visible to all waves

    v16bf a, b0, b1;
#pragma unroll
    for (int e = 0; e < 16; ++e) {
      // A (16x32 bf16): lanes 0-15 K = e / e+8 ; lanes 16-31 K = e+8 / e+16
      int kA = e + hi8 + ((e >= 8) ? 8 : 0);
      a[e]  = (__bf16)xt[kA][wv * 16 + m];
      // B (32x16 bf16): same K order as A ; N = lane%16
      int kB = kb + e + hi8 + ((e >= 8) ? 8 : 0);
      b0[e] = wl[kB * CD + m];
      b1[e] = wl[kB * CD + 16 + m];
    }
    c0 = __builtin_amdgcn_wmma_f32_16x16x32_bf16(false, a, false, b0, (short)0, c0, false, false);
    c1 = __builtin_amdgcn_wmma_f32_16x16x32_bf16(false, a, false, b1, (short)0, c1, false, false);
  }
#pragma unroll
  for (int v = 0; v < 8; ++v) {
    int row = tok0 + v + hi8;                 // C: M = v + 8*(lane>=16)
    size_t o = (size_t)row * CD;
    r[o + m]         = c0[v] + b_in[m];
    r[o + 16 + m]    = c1[v] + b_in[16 + m];
    qsum[o + m]      = 0.f;
    qsum[o + 16 + m] = 0.f;
  }
}

// ---------------------------------------------------------------------------
// Stage 1 of separable adaptive avg pool: rows H(64) -> ph.
// tmp[b, p, w, d] = mean_h r[b, h, w, d];  one wave per (b,p,w), lane = d.
// ---------------------------------------------------------------------------
__global__ __launch_bounds__(256) void k_pool_rows(
    const float* __restrict__ r, float* __restrict__ tmp, int ph)
{
  const int lane = threadIdx.x & 31;
  const int wv   = threadIdx.x >> 5;
  int t = blockIdx.x * 8 + wv;                // (b, p, w) flat
  int total = BATCH * ph * 64;
  if (t >= total) return;                     // wave-uniform exit
  int b   = t / (ph * 64);
  int rem = t - b * ph * 64;
  int p   = rem >> 6;
  int w   = rem & 63;
  int hs = (p * 64) / ph, he = ((p + 1) * 64 + ph - 1) / ph;
  float s = 0.f;
  for (int h = hs; h < he; ++h)
    s += r[((size_t)(b * HW + h * 64 + w)) * CD + lane];
  tmp[(size_t)t * CD + lane] = s / (float)(he - hs);
}

// ---------------------------------------------------------------------------
// Stage 2: pool cols W(64) -> pw, then BSQ: L2-normalize per token, sign
// quantize, accumulate loss statistics. One token per wave (lane == d).
// acc: [0..31] sum p per dim, [32] sum persample entropy, [33] sum (z-qhat)^2
// ---------------------------------------------------------------------------
__global__ __launch_bounds__(256) void k_bsq(
    const float* __restrict__ tmp, float* __restrict__ qs,
    float* __restrict__ acc, int ph, int pw)
{
  __shared__ float lp[CD];
  __shared__ float lpent;
  __shared__ float lcommit;
  const int tid = threadIdx.x;
  if (tid < CD)      lp[tid] = 0.f;
  if (tid == CD)     lpent   = 0.f;
  if (tid == CD + 1) lcommit = 0.f;
  __syncthreads();

  const int lane  = tid & 31;
  const int wv    = tid >> 5;
  const int npix  = ph * pw;
  const int total = BATCH * npix;
  int tok = blockIdx.x * 8 + wv;
  const bool valid = tok < total;
  int tokc = valid ? tok : 0;
  int b  = tokc / npix;
  int pq = tokc - b * npix;
  int p  = pq / pw;
  int q  = pq - p * pw;
  int ws = (q * 64) / pw, we = ((q + 1) * 64 + pw - 1) / pw;
  float s = 0.f;
  for (int w = ws; w < we; ++w)
    s += tmp[((size_t)((b * ph + p) * 64 + w)) * CD + lane];
  float zr = s / (float)(we - ws);

  float nrm = fmaxf(sqrtf(wave_sum(zr * zr)), 1e-12f);
  float z = zr / nrm;
  const float isd = 0.17677669529663687f;          // 1/sqrt(32)
  float arg = 4.0f * 100.0f * isd * z;             // INV_T = 100
  float pb  = 1.0f / (1.0f + __expf(arg));         // sigmoid(-arg)
  float ent = -(pb * __logf(pb + 1e-8f) + (1.f - pb) * __logf(1.f - pb + 1e-8f));
  float qh  = (z > 0.f) ? isd : -isd;
  float df  = z - qh;
  if (valid) qs[(size_t)tok * CD + lane] = qh;

  float wgt = valid ? 1.f : 0.f;
  atomicAdd(&lp[lane], pb * wgt);
  float se = wave_sum(ent * wgt);
  float sc = wave_sum(df * df * wgt);
  if (lane == 0) { atomicAdd(&lpent, se); atomicAdd(&lcommit, sc); }
  __syncthreads();
  if (tid < CD)      atomicAdd(&acc[tid],    lp[tid]);
  if (tid == CD)     atomicAdd(&acc[CD],     lpent);
  if (tid == CD + 1) atomicAdd(&acc[CD + 1], lcommit);
}

__global__ void k_zero_acc(float* __restrict__ acc) {
  if (threadIdx.x < CD + 2) acc[threadIdx.x] = 0.f;
}

__global__ void k_loss(const float* __restrict__ acc, float* __restrict__ losses,
                       int level, int total)
{
  if (threadIdx.x != 0 || blockIdx.x != 0) return;
  float inv = 1.0f / (float)total;
  float persample = acc[CD] * inv;                 // mean over (b,n) of per-d sum
  float commit    = acc[CD + 1] * inv * (1.0f / CD);
  float cb = 0.f;
  for (int d = 0; d < CD; ++d) {
    float pb = acc[d] * inv;
    float qb = 1.f - pb;
    cb += -(pb * __logf(pb + 1e-8f) + qb * __logf(qb + 1e-8f));
  }
  // (GAMMA0*persample - GAMMA*cb)/INV_T * ENT_W + COMMIT_W*commit
  losses[level] = (1.0f * persample - 1.0f * cb) * (1.0f / 100.0f) * 0.1f
                + 0.25f * commit;
}

// ---------------------------------------------------------------------------
// bilinear upsample q_small(ph,pw) -> (64,64); r -= q_up ; qsum += q_up
// half-pixel centers with edge clamp (align_corners=False)
// ---------------------------------------------------------------------------
__global__ __launch_bounds__(256) void k_up(
    const float* __restrict__ qs, float* __restrict__ r,
    float* __restrict__ qsum, int ph, int pw)
{
  const int lane = threadIdx.x & 31;
  const int wv   = threadIdx.x >> 5;
  int tok = blockIdx.x * 8 + wv;   // 0..65535
  int b  = tok >> 12;
  int hw = tok & 4095;
  int h = hw >> 6, w = hw & 63;
  float sh = fminf(fmaxf((h + 0.5f) * ((float)ph / 64.0f) - 0.5f, 0.f), (float)(ph - 1));
  int   i0 = (int)floorf(sh); int i1 = min(i0 + 1, ph - 1); float fh = sh - (float)i0;
  float sw = fminf(fmaxf((w + 0.5f) * ((float)pw / 64.0f) - 0.5f, 0.f), (float)(pw - 1));
  int   j0 = (int)floorf(sw); int j1 = min(j0 + 1, pw - 1); float fw = sw - (float)j0;
  size_t base = (size_t)b * ph * pw;
  float q00 = qs[(base + i0 * pw + j0) * CD + lane];
  float q01 = qs[(base + i0 * pw + j1) * CD + lane];
  float q10 = qs[(base + i1 * pw + j0) * CD + lane];
  float q11 = qs[(base + i1 * pw + j1) * CD + lane];
  float qv = (1.f - fh) * ((1.f - fw) * q00 + fw * q01)
           +        fh  * ((1.f - fw) * q10 + fw * q11);
  size_t idx = (size_t)tok * CD + lane;
  r[idx]    -= qv;
  qsum[idx] += qv;
}

// ---------------------------------------------------------------------------
// project_out: out[b, c, hw] = sum_k qsum[tok, k] * w_out[k, c] + b_out[c]
// K = 32 (single bf16 WMMA K-step). Block = 16 tokens x 512 channels;
// wave wv covers channels [wv*64, wv*64+64). C tiles transposed via LDS so
// the channel-major store is coalesced.
// ---------------------------------------------------------------------------
__global__ __launch_bounds__(256) void k_project_out(
    const float* __restrict__ qsum, const float* __restrict__ w_out,
    const float* __restrict__ b_out, float* __restrict__ out)
{
  __shared__ float lds[DIM][17];
  const int lane = threadIdx.x & 31;
  const int wv   = threadIdx.x >> 5;
  const int tok0 = blockIdx.x * 16;
  const int b    = tok0 >> 12;
  const int hw0  = tok0 & 4095;
  const int m    = lane & 15;
  const int hi8  = (lane >= 16) ? 8 : 0;
  const int hi16 = (lane >= 16) ? 16 : 0;

  v16bf a;
#pragma unroll
  for (int e = 0; e < 16; ++e) {
    int kA = e + hi8 + ((e >= 8) ? 8 : 0);
    a[e] = (__bf16)qsum[(size_t)(tok0 + m) * CD + kA];
  }
#pragma unroll
  for (int j = 0; j < 4; ++j) {
    int n0 = wv * 64 + j * 16;
    v16bf bb;
#pragma unroll
    for (int e = 0; e < 16; ++e) {
      int kB = e + hi8 + ((e >= 8) ? 8 : 0);
      bb[e] = (__bf16)w_out[kB * DIM + n0 + m];
    }
    v8f c = {};
    c = __builtin_amdgcn_wmma_f32_16x16x32_bf16(false, a, false, bb, (short)0, c, false, false);
#pragma unroll
    for (int v = 0; v < 8; ++v) lds[n0 + m][v + hi8] = c[v];
  }
  __syncthreads();
  for (int ch = threadIdx.x; ch < DIM; ch += 256) {
    float bias = b_out[ch];
    size_t base = ((size_t)(b * DIM + ch) << 12) + hw0;
#pragma unroll
    for (int t = 0; t < 16; ++t) out[base + t] = lds[ch][t] + bias;
  }
}

// ---------------------------------------------------------------------------
extern "C" void kernel_launch(void* const* d_in, const int* in_sizes, int n_in,
                              void* d_out, int out_size, void* d_ws, size_t ws_size,
                              hipStream_t stream)
{
  const float* x     = (const float*)d_in[0];
  const float* w_in  = (const float*)d_in[1];
  const float* b_in  = (const float*)d_in[2];
  const float* w_out = (const float*)d_in[3];
  const float* b_out = (const float*)d_in[4];
  float* out    = (float*)d_out;
  float* losses = out + (size_t)BATCH * DIM * HW;   // 14 loss scalars after out

  char*  ws   = (char*)d_ws;
  float* r    = (float*)(ws);                       //  8 MB: residual (65536 x 32)
  float* qsum = (float*)(ws + ((size_t)8  << 20));  //  8 MB: accumulated quantized
  float* qs   = (float*)(ws + ((size_t)16 << 20));  //  8 MB: per-level q
  float* tmp  = (float*)(ws + ((size_t)24 << 20));  //  8 MB: row-pooled partial
  float* acc  = (float*)(ws + ((size_t)32 << 20));  //  34 floats: loss accums

  k_project_in<<<512, 256, 0, stream>>>(x, w_in, b_in, r, qsum);

  static const int sched[14][2] = {
    {1,1},{2,2},{3,3},{4,4},{5,5},{7,7},{9,9},
    {12,12},{16,16},{21,21},{27,27},{36,36},{48,48},{64,64}};
  for (int lv = 0; lv < 14; ++lv) {
    int ph = sched[lv][0], pw = sched[lv][1];
    int total  = BATCH * ph * pw;
    int rowtot = BATCH * ph * 64;
    k_zero_acc<<<1, 64, 0, stream>>>(acc);
    k_pool_rows<<<(rowtot + 7) / 8, 256, 0, stream>>>(r, tmp, ph);
    k_bsq<<<(total + 7) / 8, 256, 0, stream>>>(tmp, qs, acc, ph, pw);
    k_loss<<<1, 1, 0, stream>>>(acc, losses, lv, total);
    k_up<<<NTOK_FULL / 8, 256, 0, stream>>>(qs, r, qsum, ph, pw);
  }

  k_project_out<<<4096, 256, 0, stream>>>(qsum, w_out, b_out, out);
}
